// BiEmbeddingToBiAGNN_69157563400718
// MI455X (gfx1250) — matrix operand order repaired and checked by reading, not compiled, not run on hardware
//
#include <hip/hip_runtime.h>
#include <hip/hip_bf16.h>

// CDNA5 (gfx1250) GNN forward. Edge MLP uses v_wmma_f32_16x16x32_f16:
//   Y^T = W^T (A-frag, M=neuron,K=input) x X^T (B-frag, K=input,N=edge)
// Activations stay in D-fragment layout between layers; LayerNorm and the
// next-layer B fragment use single shfl_xor(16) lane-pair swaps. All
// parameter access in the MLP chain is branch-free (LDS + clamp + mask).

typedef _Float16 v16h __attribute__((ext_vector_type(16)));
typedef float    v8f  __attribute__((ext_vector_type(8)));

#define ND 19   // node feature dim (8 hidden + 8 spatial + 3 x)

__device__ __forceinline__ float sigm(float x){ return 1.0f/(1.0f+__expf(-x)); }

__device__ __forceinline__ void ln_tanh8(float* v, const float* g, const float* b){
  float m=0.f;
  #pragma unroll
  for(int j=0;j<8;++j) m+=v[j];
  m*=0.125f;
  float va=0.f;
  #pragma unroll
  for(int j=0;j<8;++j){ float d=v[j]-m; va+=d*d; }
  va*=0.125f;
  float r=rsqrtf(va+1e-5f);
  #pragma unroll
  for(int j=0;j<8;++j) v[j]=tanhf((v[j]-m)*r*g[j]+b[j]);
}

// ---- zero scratch ----
__global__ void k_zero(float* __restrict__ p, int n){
  int i = blockIdx.x*blockDim.x + threadIdx.x;
  if(i<n) p[i]=0.f;
}

// ---- node embedding + input network: x[N,3] -> spatial[N,8], h[N,19] ----
__global__ __launch_bounds__(256) void k_embed(
  const float* __restrict__ x,
  const float* __restrict__ w0,const float* __restrict__ b0,
  const float* __restrict__ w1,const float* __restrict__ b1,
  const float* __restrict__ w2,const float* __restrict__ b2,
  const float* __restrict__ w3,const float* __restrict__ b3,
  const float* __restrict__ we,const float* __restrict__ be,
  const float* __restrict__ wi,const float* __restrict__ bi,
  const float* __restrict__ lg,const float* __restrict__ lb,
  float* __restrict__ h, float* __restrict__ sp_out, int n)
{
  __shared__ float s[3648];
  const float* srcs[14]={w0,b0,w1,b1,w2,b2,w3,b3,we,be,wi,bi,lg,lb};
  const int offs[15]={0,96,128,1152,1184,2208,2240,3264,3296,3552,3560,3624,3632,3640,3648};
  for(int a=0;a<14;++a){
    int cnt=offs[a+1]-offs[a];
    for(int i=threadIdx.x;i<cnt;i+=blockDim.x) s[offs[a]+i]=srcs[a][i];
  }
  __syncthreads();
  int nid = blockIdx.x*blockDim.x + threadIdx.x;
  if(nid>=n) return;
  float xv[3]; xv[0]=x[nid*3+0]; xv[1]=x[nid*3+1]; xv[2]=x[nid*3+2];
  float a0[32], a1[32];
  const float* W0=s+0; const float* B0=s+96;
  #pragma unroll 8
  for(int j=0;j<32;++j)
    a0[j]=tanhf(B0[j]+xv[0]*W0[j]+xv[1]*W0[32+j]+xv[2]*W0[64+j]);
  const int lwo[3]={128,1184,2240}, lbo[3]={1152,2208,3264};
  for(int L=0;L<3;++L){
    const float* W=s+lwo[L]; const float* B=s+lbo[L];
    for(int j=0;j<32;++j){
      float acc=B[j];
      #pragma unroll 8
      for(int i=0;i<32;++i) acc+=a0[i]*W[i*32+j];
      a1[j]=tanhf(acc);
    }
    for(int j=0;j<32;++j) a0[j]=a1[j];
  }
  float sp[8];
  const float* WE=s+3296; const float* BE=s+3552;
  #pragma unroll
  for(int j=0;j<8;++j){
    float acc=BE[j];
    #pragma unroll 8
    for(int i=0;i<32;++i) acc+=a0[i]*WE[i*8+j];
    sp[j]=acc;
  }
  float y[8];
  const float* WI=s+3560; const float* BI=s+3624;
  #pragma unroll
  for(int j=0;j<8;++j){
    float acc=BI[j];
    #pragma unroll
    for(int i=0;i<8;++i) acc+=sp[i]*WI[i*8+j];
    y[j]=acc;
  }
  ln_tanh8(y, s+3632, s+3640);
  float* hp = h + (size_t)nid*ND;
  #pragma unroll
  for(int j=0;j<8;++j) hp[j]=y[j];
  #pragma unroll
  for(int j=0;j<8;++j) hp[8+j]=sp[j];
  hp[16]=xv[0]; hp[17]=xv[1]; hp[18]=xv[2];
  #pragma unroll
  for(int j=0;j<8;++j) sp_out[(size_t)nid*8+j]=sp[j];
}

// ---- edge-MLP helpers (activations live in D-fragment layout) ----
// d[r] holds row j=8*hi+r (neuron) of this lane's edge column.
// Branch-free: params come from LDS with clamped index + mask multiply.
__device__ __forceinline__ void postln11(v8f& d, int hi,
    const float* sbias, const float* sg, const float* sbb)
{
  float msk[8];
  #pragma unroll
  for(int r=0;r<8;++r){
    int j=8*hi+r;
    int jc=(j<11)? j : 10;
    msk[r]=(j<11)? 1.f : 0.f;
    d[r]+=msk[r]*sbias[jc];
  }
  float s1=0.f;
  #pragma unroll
  for(int r=0;r<8;++r) s1+=d[r];             // invalid rows are exactly 0
  s1 += __shfl_xor(s1,16);
  float mean=s1*(1.f/11.f);
  float s2=0.f;
  #pragma unroll
  for(int r=0;r<8;++r){ float t=d[r]-mean; s2+=msk[r]*t*t; }
  s2 += __shfl_xor(s2,16);
  float rstd=rsqrtf(s2*(1.f/11.f)+1e-5f);
  #pragma unroll
  for(int r=0;r<8;++r){
    int j=8*hi+r; int jc=(j<11)? j : 10;
    d[r]=msk[r]*tanhf((d[r]-mean)*rstd*sg[jc]+sbb[jc]);
  }
}

// Build next-layer B fragment (K=inputs 0..10, N=edge) from D fragment.
// Lane hi=0 needs rows 8..10 which live in lane L+16 -> 3-value shfl swap.
__device__ __forceinline__ v16h makeB11(const v8f& d, int hi){
  float hm=(hi==0)? 1.f : 0.f;
  float o0=__shfl_xor(d[0],16);
  float o1=__shfl_xor(d[1],16);
  float o2=__shfl_xor(d[2],16);
  v16h b={};
  #pragma unroll
  for(int e=0;e<8;++e) b[e]=(_Float16)(hm*d[e]);
  b[8]=(_Float16)(hm*o0); b[9]=(_Float16)(hm*o1); b[10]=(_Float16)(hm*o2);
  return b;
}

// ---- edge pass: 16 edges per wave, 4 waves/block ----
// mode 0: accumulate mi[end]+=e*h[start], mo[start]+=e*h[end]
// mode 1: write logits
__global__ __launch_bounds__(128) void k_edge(
  const int* __restrict__ ei, int E,
  const float* __restrict__ h,
  float* __restrict__ mi, float* __restrict__ mo,
  float* __restrict__ logits,
  const float* __restrict__ w1,const float* __restrict__ b1,
  const float* __restrict__ w2,const float* __restrict__ b2,
  const float* __restrict__ w3,const float* __restrict__ b3,
  const float* __restrict__ w4,const float* __restrict__ b4,
  const float* __restrict__ g0,const float* __restrict__ lb0,
  const float* __restrict__ g1,const float* __restrict__ lb1,
  const float* __restrict__ g2,const float* __restrict__ lb2,
  int mode)
{
  // Stage all edge-MLP parameters in LDS (771 floats = ~3KB).
  __shared__ float s[771];
  {
    const float* srcs[14]={w1,w2,w3,w4,b1,b2,b3,b4,g0,lb0,g1,lb1,g2,lb2};
    const int offs[15]={0,418,539,660,671,682,693,704,705,716,727,738,749,760,771};
    for(int a=0;a<14;++a){
      int cnt=offs[a+1]-offs[a];
      for(int i=threadIdx.x;i<cnt;i+=blockDim.x) s[offs[a]+i]=srcs[a][i];
    }
  }
  __syncthreads();

  const int lane = threadIdx.x & 31;
  const int wave = threadIdx.x >> 5;
  const int hi   = lane >> 4;
  const int col  = lane & 15;
  const long base = ((long)blockIdx.x*4 + wave) * 16;
  if(base >= E) return;
  int eid = (int)base + col;
  const bool valid = (eid < E);
  const int eidc = valid ? eid : (E-1);
  const int sN = ei[eidc];
  const int tN = ei[E + eidc];
  const float* __restrict__ hs = h + (size_t)sN*ND;
  const float* __restrict__ ht = h + (size_t)tN*ND;

  const float* sw1=s;       const float* sw2=s+418;  const float* sw3=s+539;
  const float* sw4=s+660;   const float* sb1=s+671;  const float* sb2=s+682;
  const float* sb3=s+693;   const float  sb4=s[704];
  const float* sg0=s+705;   const float* sl0=s+716;
  const float* sg1=s+727;   const float* sl1=s+738;
  const float* sg2=s+749;   const float* sl2=s+760;

  // A fragments = W^T tiles (contiguous K order, matched by the B fragments):
  // lane M=col; elem e -> K = 16*hi+e
  const int   M  = col;
  const int   Mc = (M<11)? M : 10;
  const float Mm = (M<11)? 1.f : 0.f;
  v16h A0={},A1={},A2={},A3={};
  #pragma unroll
  for(int e=0;e<16;++e){
    int K  = 16*hi + e;
    int Kb = K+32;
    int Kc  = (K <38)? K  : 37;  float mK  = (K <38)? Mm : 0.f;
    int Kbc = (Kb<38)? Kb : 37;  float mKb = (Kb<38)? Mm : 0.f;
    int K11 = (K <11)? K  : 10;  float m11 = (K <11)? Mm : 0.f;
    A0[e]=(_Float16)(mK *sw1[Kc *11+Mc]);
    A1[e]=(_Float16)(mKb*sw1[Kbc*11+Mc]);
    A2[e]=(_Float16)(m11*sw2[K11*11+Mc]);
    A3[e]=(_Float16)(m11*sw3[K11*11+Mc]);
  }

  // B fragments = gathered edge features f = [h[s](19) | h[t](19) | pad]
  // B 32x16 layout: lane N=col; elem e -> K = 16*hi+e
  v16h B0={},B1={};
  #pragma unroll
  for(int e=0;e<16;++e){
    int K = 16*hi + e;
    float f = (K<19)? hs[K] : ((K<38)? ht[K-19] : 0.f);
    B0[e]=(_Float16)f;
    int K2 = K + 32;                       // 32..63
    float f2 = (K2<38)? ht[K2-19] : 0.f;   // only 32..37 valid
    B1[e]=(_Float16)f2;
  }

  // Layer 1: 38->11 (K padded to 64 => 2 WMMAs)
  v8f c = {};
  c = __builtin_amdgcn_wmma_f32_16x16x32_f16(false, A0, false, B0, (short)0, c, false, false);
  c = __builtin_amdgcn_wmma_f32_16x16x32_f16(false, A1, false, B1, (short)0, c, false, false);
  postln11(c, hi, sb1, sg0, sl0);

  // Layer 2: 11->11
  v16h Bx = makeB11(c, hi);
  v8f c2 = {};
  c2 = __builtin_amdgcn_wmma_f32_16x16x32_f16(false, A2, false, Bx, (short)0, c2, false, false);
  postln11(c2, hi, sb2, sg1, sl1);

  // Layer 3: 11->11
  Bx = makeB11(c2, hi);
  v8f c3 = {};
  c3 = __builtin_amdgcn_wmma_f32_16x16x32_f16(false, A3, false, Bx, (short)0, c3, false, false);
  postln11(c3, hi, sb3, sg2, sl2);

  // Final layer 11->1: lane-pair dot product (mask-multiplied, branch-free)
  float part=0.f;
  #pragma unroll
  for(int r=0;r<8;++r){
    int j=8*hi+r; int jc=(j<11)? j : 10; float mk=(j<11)? 1.f : 0.f;
    part += mk*c3[r]*sw4[jc];
  }
  part += __shfl_xor(part,16);
  const float logit = part + sb4;

  if(mode==1){
    if(valid && hi==0) logits[eid]=logit;
  }else{
    const float ew = sigm(logit);
    if(valid){
      if(hi==0){
        float* dst = mi + (size_t)tN*ND;
        #pragma unroll
        for(int k=0;k<ND;++k) atomicAdd(dst+k, ew*hs[k]);
      }else{
        float* dst = mo + (size_t)sN*ND;
        #pragma unroll
        for(int k=0;k<ND;++k) atomicAdd(dst+k, ew*ht[k]);
      }
    }
  }
}

// ---- node update: h = h0 + [nodeMLP(mi|mo|h), spatial, x] ----
__global__ __launch_bounds__(256) void k_node(
  const float* __restrict__ mi, const float* __restrict__ mo,
  float* __restrict__ h, const float* __restrict__ sp, const float* __restrict__ x,
  const float* w0,const float* b0,const float* w1,const float* b1,
  const float* w2,const float* b2,const float* w3,const float* b3,
  const float* g0,const float* lb0,const float* g1,const float* lb1,
  const float* g2,const float* lb2, int n)
{
  __shared__ float s[728];
  const float* srcs[14]={w0,w1,w2,w3,b0,b1,b2,b3,g0,lb0,g1,lb1,g2,lb2};
  const int offs[15]={0,456,520,584,648,656,664,672,680,688,696,704,712,720,728};
  for(int a=0;a<14;++a){
    int cnt=offs[a+1]-offs[a];
    for(int i=threadIdx.x;i<cnt;i+=blockDim.x) s[offs[a]+i]=srcs[a][i];
  }
  __syncthreads();
  int nid = blockIdx.x*blockDim.x + threadIdx.x;
  if(nid>=n) return;
  float in[57], hold[ND];
  #pragma unroll 4
  for(int k=0;k<ND;++k) in[k]    = mi[(size_t)nid*ND+k];
  #pragma unroll 4
  for(int k=0;k<ND;++k) in[19+k] = mo[(size_t)nid*ND+k];
  #pragma unroll 4
  for(int k=0;k<ND;++k){ hold[k]=h[(size_t)nid*ND+k]; in[38+k]=hold[k]; }
  float t0[8];
  #pragma unroll
  for(int j=0;j<8;++j){
    float acc=s[648+j];
    for(int i=0;i<57;++i) acc+=in[i]*s[i*8+j];
    t0[j]=acc;
  }
  ln_tanh8(t0, s+680, s+688);
  float t1[8];
  #pragma unroll
  for(int j=0;j<8;++j){
    float acc=s[656+j];
    #pragma unroll
    for(int i=0;i<8;++i) acc+=t0[i]*s[456+i*8+j];
    t1[j]=acc;
  }
  ln_tanh8(t1, s+696, s+704);
  float t2[8];
  #pragma unroll
  for(int j=0;j<8;++j){
    float acc=s[664+j];
    #pragma unroll
    for(int i=0;i<8;++i) acc+=t1[i]*s[520+i*8+j];
    t2[j]=acc;
  }
  ln_tanh8(t2, s+712, s+720);
  float t3[8];
  #pragma unroll
  for(int j=0;j<8;++j){
    float acc=s[672+j];
    #pragma unroll
    for(int i=0;i<8;++i) acc+=t2[i]*s[584+i*8+j];
    t3[j]=acc;
  }
  float* hp = h + (size_t)nid*ND;
  #pragma unroll
  for(int k=0;k<8;++k) hp[k]   = hold[k]   + t3[k];
  #pragma unroll
  for(int k=0;k<8;++k) hp[8+k] = hold[8+k] + sp[(size_t)nid*8+k];
  hp[16]=hold[16]+x[nid*3+0];
  hp[17]=hold[17]+x[nid*3+1];
  hp[18]=hold[18]+x[nid*3+2];
}

// Input index map (jax tree_leaves order: edge_index, params{edge,emb,inp,node}, x;
// within dicts keys sorted, 'b' < 'g' < 'w', 'emb' < 'layers', 'lin' < 'ln'):
//  0 edge_index | 1..8 edge lin{b,w}x4 | 9..14 edge ln{b,g}x3
// 15 emb.emb.b 16 emb.emb.w | 17..24 emb.layers{b,w}x4
// 25 inp.lin.b 26 inp.lin.w 27 inp.ln.b 28 inp.ln.g
// 29..36 node lin{b,w}x4 | 37..42 node ln{b,g}x3 | 43 x
extern "C" void kernel_launch(void* const* d_in, const int* in_sizes, int n_in,
                              void* d_out, int out_size, void* d_ws, size_t ws_size,
                              hipStream_t stream)
{
  const int* ei = (const int*)d_in[0];
  auto F = [&](int i){ return (const float*)d_in[i]; };
  const int E = in_sizes[0]/2;
  const int N = in_sizes[43]/3;

  float* out    = (float*)d_out;
  float* logits = out;
  float* sp_out = out + E;

  float* h  = (float*)d_ws;
  float* mi = h  + (size_t)N*ND;
  float* mo = mi + (size_t)N*ND;

  const int nblkN = (N+255)/256;
  const int nblkE = (int)(((long)E+63)/64);
  const int nzero = 2*N*ND;

  k_embed<<<nblkN,256,0,stream>>>(F(43),
      F(18),F(17), F(20),F(19), F(22),F(21), F(24),F(23),
      F(16),F(15), F(26),F(25), F(28),F(27),
      h, sp_out, N);

  for(int it=0; it<3; ++it){
    k_zero<<<(nzero+255)/256,256,0,stream>>>(mi, nzero);
    k_edge<<<nblkE,128,0,stream>>>(ei,E,h,mi,mo,logits,
        F(2),F(1), F(4),F(3), F(6),F(5), F(8),F(7),
        F(10),F(9), F(12),F(11), F(14),F(13), 0);
    k_node<<<nblkN,256,0,stream>>>(mi,mo,h,sp_out,F(43),
        F(30),F(29), F(32),F(31), F(34),F(33), F(36),F(35),
        F(38),F(37), F(40),F(39), F(42),F(41), N);
  }
  k_edge<<<nblkE,128,0,stream>>>(ei,E,h,mi,mo,logits,
      F(2),F(1), F(4),F(3), F(6),F(5), F(8),F(7),
      F(10),F(9), F(12),F(11), F(14),F(13), 1);
}
